// SelfAttention_39350490366717
// MI455X (gfx1250) — matrix operand order repaired and checked by reading, not compiled, not run on hardware
//
#include <hip/hip_runtime.h>
#include <hip/hip_bf16.h>

typedef __attribute__((ext_vector_type(16))) _Float16 v16h;
typedef __attribute__((ext_vector_type(8)))  float    v8f;
typedef __attribute__((ext_vector_type(4)))  float    f32x4;

// Load a 16x32 f16 A-fragment's per-lane data: two contiguous 16B chunks.
__device__ __forceinline__ v16h load2x8h(const _Float16* p0, const _Float16* p1) {
    union { v16h v; f32x4 q[2]; } u;
    u.q[0] = *reinterpret_cast<const f32x4*>(p0);
    u.q[1] = *reinterpret_cast<const f32x4*>(p1);
    return u.v;
}

// CDNA5 async copy: global -> LDS, 16 bytes per lane, tracked by ASYNCcnt.
__device__ __forceinline__ void async_ld_b128(unsigned lds_off, const void* gaddr) {
    asm volatile("global_load_async_to_lds_b128 %0, %1, off"
                 :: "v"(lds_off), "v"(gaddr) : "memory");
}
__device__ __forceinline__ void wait_asynccnt0() {
    asm volatile("s_wait_asynccnt 0x0" ::: "memory");
}

// ---------------- LayerNorm: fp32 x -> f16 xn -------------------------------
__global__ __launch_bounds__(256) void ln_kernel(const float* __restrict__ x,
                                                 const float* __restrict__ g,
                                                 const float* __restrict__ bb,
                                                 _Float16* __restrict__ xn, int D) {
    __shared__ float red[256];
    const int tid = threadIdx.x;
    const size_t row = blockIdx.x;
    const float* xr = x + row * (size_t)D;
    float s = 0.f;
    for (int i = tid; i < D; i += 256) s += xr[i];
    red[tid] = s; __syncthreads();
    for (int st = 128; st; st >>= 1) { if (tid < st) red[tid] += red[tid + st]; __syncthreads(); }
    const float mean = red[0] / (float)D;
    __syncthreads();
    float vv = 0.f;
    for (int i = tid; i < D; i += 256) { float d = xr[i] - mean; vv += d * d; }
    red[tid] = vv; __syncthreads();
    for (int st = 128; st; st >>= 1) { if (tid < st) red[tid] += red[tid + st]; __syncthreads(); }
    const float rstd = rsqrtf(red[0] / (float)D + 1e-5f);
    for (int i = tid; i < D; i += 256)
        xn[row * (size_t)D + i] = (_Float16)((xr[i] - mean) * rstd * g[i] + bb[i]);
}

// ---------------- Weight transpose + fp32->f16 ------------------------------
// WT[n*K + k] = W[k*N + n]
__global__ __launch_bounds__(256) void wt_kernel(const float* __restrict__ W,
                                                 _Float16* __restrict__ WT,
                                                 int Kd, int Nd) {
    size_t idx = (size_t)blockIdx.x * 256 + threadIdx.x;
    int n = (int)(idx % Nd);
    int k = (int)(idx / Nd);
    WT[(size_t)n * Kd + k] = (_Float16)W[idx];
}

// ---------------- QKV GEMM: C[M,N] = A[M,K] * W, B given transposed ---------
__global__ __launch_bounds__(128) void gemm_qkv(const _Float16* __restrict__ A,
                                                const _Float16* __restrict__ BT,
                                                const float* __restrict__ bias,
                                                _Float16* __restrict__ C,
                                                int M, int N, int K) {
    const int lane = threadIdx.x & 31;
    const int wave = blockIdx.x * 4 + (threadIdx.x >> 5);
    const int half = lane >> 4, l16 = lane & 15;
    const int tilesN = N / 64;
    const int m0 = (wave / tilesN) * 64;
    const int n0 = (wave % tilesN) * 64;

    v8f acc[4][4] = {};
    for (int k0 = 0; k0 < K; k0 += 32) {
        v16h af[4], bf[4];
#pragma unroll
        for (int i = 0; i < 4; ++i) {
            const _Float16* ar = A + (size_t)(m0 + i * 16 + l16) * K + k0 + half * 8;
            af[i] = load2x8h(ar, ar + 16);
            const _Float16* br = BT + (size_t)(n0 + i * 16 + l16) * K + k0 + half * 8;
            bf[i] = load2x8h(br, br + 16);
        }
#pragma unroll
        for (int i = 0; i < 4; ++i)
#pragma unroll
            for (int j = 0; j < 4; ++j)
                acc[i][j] = __builtin_amdgcn_wmma_f32_16x16x32_f16(
                    false, af[i], false, bf[j], (short)0, acc[i][j], false, false);
    }
#pragma unroll
    for (int i = 0; i < 4; ++i)
#pragma unroll
        for (int j = 0; j < 4; ++j) {
            const int col = n0 + j * 16 + l16;
            const float bb = bias[col];
#pragma unroll
            for (int v = 0; v < 8; ++v) {
                const int row = m0 + i * 16 + v + half * 8;
                C[(size_t)row * N + col] = (_Float16)(acc[i][j][v] + bb);
            }
        }
}

// ---------------- V transpose per head: vt[b,h,dh,t] = v[b,t,h,dh] ----------
__global__ __launch_bounds__(256) void vtrans(const _Float16* __restrict__ v,
                                              _Float16* __restrict__ vt,
                                              int T, int Hn, int Dh) {
    const int D = Hn * Dh;
    size_t idx = (size_t)blockIdx.x * 256 + threadIdx.x;   // over B*T*D
    int d = (int)(idx % D);
    size_t bt = idx / D;
    int t = (int)(bt % T);
    size_t b = bt / T;
    int h = d / Dh, dh = d % Dh;
    vt[((b * Hn + h) * (size_t)Dh + dh) * T + t] = v[idx];
}

// ---------------- Flash attention -------------------------------------------
// Block = 4 waves, all sharing (b,h); each wave owns 16 query rows.
// K/V chunks (32 keys) staged once per block into LDS via async-to-LDS copies.
#define KPAD 72   // K tile row stride in f16 (64 data + 8 pad)
#define VPAD 40   // V tile row stride in f16 (32 data + 8 pad)
__global__ __launch_bounds__(128) void attn_kernel(const _Float16* __restrict__ Q,
                                                   const _Float16* __restrict__ Km,
                                                   const _Float16* __restrict__ VT,
                                                   float* __restrict__ O,
                                                   int T, int Hn, int Dh) {
    const int lane = threadIdx.x & 31, widx = threadIdx.x >> 5;
    const int half = lane >> 4, l16 = lane & 15;
    const int wid  = blockIdx.x * 4 + widx;
    const int qbPerSeq = T / 16;
    const int qblk = wid % qbPerSeq;
    const int h    = (wid / qbPerSeq) % Hn;
    const int b    = wid / (qbPerSeq * Hn);
    const int D    = Hn * Dh;
    const int m0   = qblk * 16;

    __shared__ _Float16 Ktile[32 * KPAD];       // [key][dim], padded rows
    __shared__ _Float16 Vtile[64 * VPAD];       // [dim][key], padded rows
    __shared__ float    Sbuf[4][16 * 33];
    __shared__ _Float16 Pbuf[4][16 * 32];
    __shared__ float    corrLds[4][16];
    __shared__ float    statLds[4][16];
    float*    S    = Sbuf[widx];
    _Float16* P    = Pbuf[widx];
    float*    corr = corrLds[widx];
    float*    stat = statLds[widx];

    const _Float16 hscale = (_Float16)rsqrtf((float)Dh);   // 1/8 (exact in f16)

    // Q fragments (16 queries x 64 dims -> two 16x32 A-frags), pre-scaled
    v16h qf[2];
    {
        const _Float16* qr = Q + (size_t)(b * T + m0 + l16) * D + h * Dh;
#pragma unroll
        for (int kk = 0; kk < 2; ++kk) {
            v16h t = load2x8h(qr + kk * 32 + half * 8, qr + kk * 32 + 16 + half * 8);
#pragma unroll
            for (int e = 0; e < 16; ++e) t[e] = t[e] * hscale;
            qf[kk] = t;
        }
    }

    v8f o[4] = {};
    float mrun = -1e30f, lrun = 0.f;

    // Block-uniform bases (all 4 waves share b,h).
    const _Float16* Kbase = Km + (size_t)b * T * D + h * Dh;
    const _Float16* Vbase = VT + ((size_t)b * Hn + h) * Dh * (size_t)T;

    // Per-thread staging assignments (fixed across iterations).
    const int krow = threadIdx.x >> 2, kseg = threadIdx.x & 3;   // 32 rows x 4 segs
    const int vrow = threadIdx.x >> 1, vseg = threadIdx.x & 1;   // 64 rows x 2 segs
    const unsigned kldsOff = (unsigned)(uintptr_t)&Ktile[krow * KPAD + kseg * 16];
    const unsigned vldsOff = (unsigned)(uintptr_t)&Vtile[vrow * VPAD + vseg * 16];

    for (int kc = 0; kc < T; kc += 32) {
        // ---- async stage K chunk (32x64 f16) and V chunk (64x32 f16) ----
        {
            const _Float16* gk = Kbase + (size_t)(kc + krow) * D + kseg * 16;
            async_ld_b128(kldsOff,      gk);
            async_ld_b128(kldsOff + 16, gk + 8);
            const _Float16* gv = Vbase + (size_t)vrow * T + kc + vseg * 16;
            async_ld_b128(vldsOff,      gv);
            async_ld_b128(vldsOff + 16, gv + 8);
            wait_asynccnt0();
        }
        __syncthreads();

        // ---- S tile: 16 queries x 32 keys (scale pre-folded into Q) ----
        v8f s0 = {}, s1 = {};
#pragma unroll
        for (int kk = 0; kk < 2; ++kk) {
            v16h bk0 = load2x8h(&Ktile[(l16)      * KPAD + kk * 32 + half * 8],
                                &Ktile[(l16)      * KPAD + kk * 32 + 16 + half * 8]);
            v16h bk1 = load2x8h(&Ktile[(16 + l16) * KPAD + kk * 32 + half * 8],
                                &Ktile[(16 + l16) * KPAD + kk * 32 + 16 + half * 8]);
            s0 = __builtin_amdgcn_wmma_f32_16x16x32_f16(false, qf[kk], false, bk0,
                                                        (short)0, s0, false, false);
            s1 = __builtin_amdgcn_wmma_f32_16x16x32_f16(false, qf[kk], false, bk1,
                                                        (short)0, s1, false, false);
        }
#pragma unroll
        for (int v = 0; v < 8; ++v) {
            const int r = v + half * 8;
            S[r * 33 + l16]      = s0[v];
            S[r * 33 + 16 + l16] = s1[v];
        }
        __syncthreads();

        // ---- online softmax: lane r (<16) owns query row r ----
        if (lane < 16) {
            const int r = lane;
            float sv[32];
            float cmax = -1e30f;
#pragma unroll
            for (int c = 0; c < 32; ++c) { sv[c] = S[r * 33 + c]; cmax = fmaxf(cmax, sv[c]); }
            const float newm = fmaxf(mrun, cmax);
            const float cr   = __expf(mrun - newm);
            float ssum = 0.f;
#pragma unroll
            for (int c = 0; c < 32; ++c) {
                const float e = __expf(sv[c] - newm);
                ssum += e;
                P[r * 32 + c] = (_Float16)e;
            }
            lrun = lrun * cr + ssum;
            mrun = newm;
            corr[r] = cr;
        }
        __syncthreads();

        // ---- rescale accumulated O by correction factor ----
        float cf[8];
#pragma unroll
        for (int v = 0; v < 8; ++v) cf[v] = corr[v + half * 8];
#pragma unroll
        for (int f = 0; f < 4; ++f)
#pragma unroll
            for (int v = 0; v < 8; ++v) o[f][v] *= cf[v];

        // ---- O += P * V  (one A-frag from LDS, four B-frags from LDS) ----
        const _Float16* pr = P + l16 * 32;
        v16h pf = load2x8h(pr + half * 8, pr + 16 + half * 8);
#pragma unroll
        for (int f = 0; f < 4; ++f) {
            v16h bv = load2x8h(&Vtile[(f * 16 + l16) * VPAD + half * 8],
                               &Vtile[(f * 16 + l16) * VPAD + 16 + half * 8]);
            o[f] = __builtin_amdgcn_wmma_f32_16x16x32_f16(false, pf, false, bv,
                                                          (short)0, o[f], false, false);
        }
        __syncthreads();   // protect K/V tiles before next staging
    }

    if (lane < 16) stat[lane] = 1.0f / lrun;
    __syncthreads();
    float li[8];
#pragma unroll
    for (int v = 0; v < 8; ++v) li[v] = stat[v + half * 8];
#pragma unroll
    for (int f = 0; f < 4; ++f)
#pragma unroll
        for (int v = 0; v < 8; ++v) {
            const int row = m0 + v + half * 8;
            O[(size_t)(b * T + row) * D + h * Dh + f * 16 + l16] = o[f][v] * li[v];
        }
}

// ---------------- Launch -----------------------------------------------------
extern "C" void kernel_launch(void* const* d_in, const int* in_sizes, int n_in,
                              void* d_out, int out_size, void* d_ws, size_t ws_size,
                              hipStream_t stream) {
    const float* x    = (const float*)d_in[0];
    const float* ln_g = (const float*)d_in[1];
    const float* ln_b = (const float*)d_in[2];
    const float* Wq   = (const float*)d_in[3];
    const float* bq   = (const float*)d_in[4];
    const float* Wk   = (const float*)d_in[5];
    const float* bk   = (const float*)d_in[6];
    const float* Wv   = (const float*)d_in[7];
    const float* bv   = (const float*)d_in[8];
    float* out = (float*)d_out;

    const int B = 2, T = 2048, D = 1024, Hn = 16, Dh = 64;
    const int M = B * T;

    char* ws = (char*)d_ws;
    _Float16* xn  = (_Float16*)ws; ws += (size_t)M * D * sizeof(_Float16);
    _Float16* WTq = (_Float16*)ws; ws += (size_t)D * D * sizeof(_Float16);
    _Float16* WTk = (_Float16*)ws; ws += (size_t)D * D * sizeof(_Float16);
    _Float16* WTv = (_Float16*)ws; ws += (size_t)D * D * sizeof(_Float16);
    _Float16* qb  = (_Float16*)ws; ws += (size_t)M * D * sizeof(_Float16);
    _Float16* kb  = (_Float16*)ws; ws += (size_t)M * D * sizeof(_Float16);
    _Float16* vb  = (_Float16*)ws; ws += (size_t)M * D * sizeof(_Float16);
    _Float16* vt  = (_Float16*)ws; ws += (size_t)M * D * sizeof(_Float16);

    ln_kernel<<<M, 256, 0, stream>>>(x, ln_g, ln_b, xn, D);

    const int wtGrid = (D * D) / 256;
    wt_kernel<<<wtGrid, 256, 0, stream>>>(Wq, WTq, D, D);
    wt_kernel<<<wtGrid, 256, 0, stream>>>(Wk, WTk, D, D);
    wt_kernel<<<wtGrid, 256, 0, stream>>>(Wv, WTv, D, D);

    const int gemmGrid = (M / 64) * (D / 64) / 4;   // 4 waves per block
    gemm_qkv<<<gemmGrid, 128, 0, stream>>>(xn, WTq, bq, qb, M, D, D);
    gemm_qkv<<<gemmGrid, 128, 0, stream>>>(xn, WTk, bk, kb, M, D, D);
    gemm_qkv<<<gemmGrid, 128, 0, stream>>>(xn, WTv, bv, vb, M, D, D);

    vtrans<<<(M * D) / 256, 256, 0, stream>>>(vb, vt, T, Hn, Dh);

    const int attnGrid = (B * Hn * (T / 16)) / 4;   // 4 waves per block
    attn_kernel<<<attnGrid, 128, 0, stream>>>(qb, kb, vt, out, T, Hn, Dh);
}
